// MultiHeadSelfAttention_47141561041114
// MI455X (gfx1250) — matrix operand order repaired and checked by reading, not compiled, not run on hardware
//
#include <hip/hip_runtime.h>

#define SEQ   2048
#define EMB   768
#define NH    12
#define HD    64
#define ROWS  (2 * SEQ)     // 4096 total (b,s) rows
#define QKVC  (3 * EMB)     // 2304

typedef __attribute__((ext_vector_type(16))) _Float16 v16h;
typedef __attribute__((ext_vector_type(8)))  _Float16 v8h;
typedef __attribute__((ext_vector_type(8)))  float    v8f;

// ---------------------------------------------------------------- WMMA core
static __device__ __forceinline__ v8f wmma16(v16h a, v16h b, v8f c) {
  // D = A(16x32 f16) * B(32x16 f16) + C(16x16 f32)
  return __builtin_amdgcn_wmma_f32_16x16x32_f16(
      /*neg_a=*/false, a, /*neg_b=*/false, b,
      /*c_mod=*/(short)0, c, /*reuse_a=*/false, /*reuse_b=*/false);
}

// 16x32 f16 A-fragment. `row` already points at this lane's row (M=lane&15).
// kbA = (lane>=16)?8:0: lanes 0-15 hold K={0..15},
// lanes 16-31 hold K={16..31}. Works for global or LDS pointers.
static __device__ __forceinline__ v16h load_a32(const _Float16* row, int kbA) {
  v8h p0 = *(const v8h*)(row + 2 * kbA);
  v8h p1 = *(const v8h*)(row + 2 * kbA + 8);
  v16h a;
#pragma unroll
  for (int i = 0; i < 8; ++i) { a[i] = p0[i]; a[8 + i] = p1[i]; }
  return a;
}

// ------------------------------------------------------------- prep kernels
__global__ void cvt_f16_kernel(const float* __restrict__ src,
                               _Float16* __restrict__ dst, int n) {
  int i = blockIdx.x * blockDim.x + threadIdx.x;
  if (i < n) dst[i] = (_Float16)src[i];
}

// src row-major [Kdim, Ndim] f32  ->  dst row-major [Ndim, Kdim] f16
__global__ void cvt_tr_f16_kernel(const float* __restrict__ src,
                                  _Float16* __restrict__ dst, int Kdim, int Ndim) {
  int i = blockIdx.x * blockDim.x + threadIdx.x;
  if (i < Kdim * Ndim) {
    int c = i / Kdim;
    int k = i - c * Kdim;
    dst[i] = (_Float16)src[(size_t)k * Ndim + c];
  }
}

// ---------------------------------------------------- kernel 1: QKV projection
// 64x64 output tile per wave (4x4 WMMA register blocking: A frags reused over
// 4 column tiles, B frags over 4 row tiles -> 32 FLOP/byte of fragment reads).
// Scatter to Q[B,H,S,D] (pre-scaled 1/sqrt(D)), K[B,H,S,D], Vt[B,H,D,S].
// A 64-aligned column group lies in exactly one of q/k/v and one head, so the
// scatter branch is wave-uniform.
__global__ void __launch_bounds__(128)
qkv_proj_kernel(const _Float16* __restrict__ Xh, const _Float16* __restrict__ Wt,
                const float* __restrict__ bias, _Float16* __restrict__ Qh,
                _Float16* __restrict__ Kh, _Float16* __restrict__ Vth) {
  const int NT4 = QKVC / 64;                      // 36 column groups
  int wave = blockIdx.x * 4 + (threadIdx.x >> 5); // 64*36 = 2304 waves
  int mt4 = wave / NT4, nt4 = wave - mt4 * NT4;
  int lane = threadIdx.x & 31, lo = lane & 15, hb = lane >> 4;
  int kbA = hb * 8, kbB = hb * 16;
  const _Float16* arow = Xh + (size_t)(mt4 * 64 + lo) * EMB;
  const _Float16* bcol = Wt + (size_t)(nt4 * 64 + lo) * EMB;
  v8f zero = {};
  v8f acc[4][4];
#pragma unroll
  for (int mi = 0; mi < 4; ++mi)
#pragma unroll
    for (int ni = 0; ni < 4; ++ni) acc[mi][ni] = zero;

  for (int k0 = 0; k0 < EMB; k0 += 32) {
    v16h a[4], b[4];
#pragma unroll
    for (int mi = 0; mi < 4; ++mi)
      a[mi] = load_a32(arow + (size_t)mi * 16 * EMB + k0, kbA);
#pragma unroll
    for (int ni = 0; ni < 4; ++ni)
      b[ni] = *(const v16h*)(bcol + (size_t)ni * 16 * EMB + k0 + kbB);
#pragma unroll
    for (int mi = 0; mi < 4; ++mi)
#pragma unroll
      for (int ni = 0; ni < 4; ++ni)
        acc[mi][ni] = wmma16(a[mi], b[ni], acc[mi][ni]);
  }

  int cgrp = nt4 * 64;
  int sel = cgrp / EMB, rem = cgrp - sel * EMB;   // wave-uniform
  int h = rem >> 6;                               // head for whole 64-col group
  float bv[4];
#pragma unroll
  for (int ni = 0; ni < 4; ++ni) bv[ni] = bias[cgrp + ni * 16 + lo];

#pragma unroll
  for (int mi = 0; mi < 4; ++mi) {
#pragma unroll
    for (int j = 0; j < 8; ++j) {                 // C/D row = j + 8*(lane>=16)
      int R = mt4 * 64 + mi * 16 + j + 8 * hb;
      int b2 = R >> 11, s = R & (SEQ - 1);
      size_t base  = ((size_t)(b2 * NH + h) * SEQ + s) * HD;
      size_t vbase = ((size_t)(b2 * NH + h) * HD) * SEQ + s;
      if (sel == 0) {
#pragma unroll
        for (int ni = 0; ni < 4; ++ni)
          Qh[base + ni * 16 + lo] = (_Float16)((acc[mi][ni][j] + bv[ni]) * 0.125f);
      } else if (sel == 1) {
#pragma unroll
        for (int ni = 0; ni < 4; ++ni)
          Kh[base + ni * 16 + lo] = (_Float16)(acc[mi][ni][j] + bv[ni]);
      } else {
#pragma unroll
        for (int ni = 0; ni < 4; ++ni)
          Vth[vbase + (size_t)(ni * 16 + lo) * SEQ] = (_Float16)(acc[mi][ni][j] + bv[ni]);
      }
    }
  }
}

// ------------------------------------------- kernel 2: streaming attention
// Block = 4 waves = one (b,h) x 128 q rows; each wave owns 32 q rows (2 row
// tiles). Per 32-key step the block stages K(32x64) and V^T(64x32) tiles in
// LDS once (cooperative b128 copies), so K/V global traffic drops 8x vs
// one-tile-per-wave. Online softmax; P goes C-layout -> LDS -> A-layout.
__global__ void __launch_bounds__(128)
attn_kernel(const _Float16* __restrict__ Qh, const _Float16* __restrict__ Kh,
            const _Float16* __restrict__ Vth, _Float16* __restrict__ Attn) {
  __shared__ __align__(32) _Float16 Klds[32 * HD];      // [key][d]      4KB
  __shared__ __align__(32) _Float16 Vlds[HD * 32];      // [d][key]      4KB
  __shared__ __align__(32) _Float16 Plds[4][32 * 32];   // per-wave P    8KB
  int tid = threadIdx.x;
  int wv = tid >> 5;
  int bh = blockIdx.x >> 4;                       // 24 (b,h) pairs
  int qb = blockIdx.x & 15;                       // 16 q blocks of 128 rows
  int lane = tid & 31, lo = lane & 15, hb = lane >> 4;
  int kbA = hb * 8, kbB = hb * 16;
  const _Float16* Qb = Qh  + (size_t)bh * SEQ * HD;
  const _Float16* Kb = Kh  + (size_t)bh * SEQ * HD;
  const _Float16* Vb = Vth + (size_t)bh * HD * SEQ;
  int qr = qb * 128 + wv * 32;

  v16h qa[2][2];                                  // Q 32x64 held in registers
#pragma unroll
  for (int rt = 0; rt < 2; ++rt) {
    const _Float16* qrow = Qb + (size_t)(qr + rt * 16 + lo) * HD;
    qa[rt][0] = load_a32(qrow, kbA);
    qa[rt][1] = load_a32(qrow + 32, kbA);
  }
  v8f zero = {};
  v8f O[2][4];
  float m[2][8], l[2][8];
#pragma unroll
  for (int rt = 0; rt < 2; ++rt) {
#pragma unroll
    for (int t = 0; t < 4; ++t) O[rt][t] = zero;
#pragma unroll
    for (int j = 0; j < 8; ++j) { m[rt][j] = -3.0e38f; l[rt][j] = 0.f; }
  }

  int kkey = tid >> 2, kch = (tid & 3) * 16;      // K staging: 128 x b128
  int vd   = tid >> 1, vch = (tid & 1) * 16;      // V staging: 128 x b128

  for (int kk = 0; kk < SEQ; kk += 32) {
    __syncthreads();                              // Klds/Vlds readers done
    *(v16h*)(&Klds[kkey * HD + kch]) =
        *(const v16h*)(Kb + (size_t)(kk + kkey) * HD + kch);
    *(v16h*)(&Vlds[vd * 32 + vch]) =
        *(const v16h*)(Vb + (size_t)vd * SEQ + kk + vch);
    if (kk + 32 < SEQ) {                          // uniform
      __builtin_prefetch(Kb + (size_t)(kk + 32 + kkey) * HD + kch, 0, 1);
      __builtin_prefetch(Vb + (size_t)vd * SEQ + kk + 32 + vch, 0, 1);
    }
    __syncthreads();                              // staging visible

    // scores: S = Q(32x64) @ K^T(64x32); K fragments shared across row tiles
    v16h b00 = *(const v16h*)(&Klds[lo * HD + kbB]);
    v16h b01 = *(const v16h*)(&Klds[lo * HD + kbB + 32]);
    v16h b10 = *(const v16h*)(&Klds[(16 + lo) * HD + kbB]);
    v16h b11 = *(const v16h*)(&Klds[(16 + lo) * HD + kbB + 32]);
#pragma unroll
    for (int rt = 0; rt < 2; ++rt) {
      v8f c0 = zero, c1 = zero;
      c0 = wmma16(qa[rt][0], b00, c0);
      c0 = wmma16(qa[rt][1], b01, c0);
      c1 = wmma16(qa[rt][0], b10, c1);
      c1 = wmma16(qa[rt][1], b11, c1);
      // online softmax: row r = j + 8*(lane>=16) lives in one 16-lane half,
      // xor masks 1,2,4,8 reduce per row.
#pragma unroll
      for (int j = 0; j < 8; ++j) {
        float s0 = c0[j], s1 = c1[j];
        float mx = fmaxf(s0, s1);
#pragma unroll
        for (int off = 1; off < 16; off <<= 1)
          mx = fmaxf(mx, __shfl_xor(mx, off, 32));
        float mnew = fmaxf(m[rt][j], mx);
        float alpha = __expf(m[rt][j] - mnew);
        float p0 = __expf(s0 - mnew);
        float p1 = __expf(s1 - mnew);
        float ps = p0 + p1;
#pragma unroll
        for (int off = 1; off < 16; off <<= 1)
          ps += __shfl_xor(ps, off, 32);
        l[rt][j] = l[rt][j] * alpha + ps;
        m[rt][j] = mnew;
#pragma unroll
        for (int t = 0; t < 4; ++t) O[rt][t][j] *= alpha;
        Plds[wv][(rt * 16 + j + 8 * hb) * 32 + lo]      = (_Float16)p0;
        Plds[wv][(rt * 16 + j + 8 * hb) * 32 + 16 + lo] = (_Float16)p1;
      }
    }
    asm volatile("s_wait_dscnt 0" ::: "memory");  // same-wave cross-lane via LDS
    v16h pa0 = load_a32(&Plds[wv][lo * 32], kbA);
    v16h pa1 = load_a32(&Plds[wv][(16 + lo) * 32], kbA);
    // O += P(32x32) @ V(32x64); V fragments shared across row tiles
#pragma unroll
    for (int t = 0; t < 4; ++t) {
      v16h bvf = *(const v16h*)(&Vlds[(t * 16 + lo) * 32 + kbB]);
      O[0][t] = wmma16(pa0, bvf, O[0][t]);
      O[1][t] = wmma16(pa1, bvf, O[1][t]);
    }
  }

  // epilogue: O / l -> f16 attn [B,S,E] (E offset = h*64 + d)
  int b2 = bh / NH, h = bh - b2 * NH;
  _Float16* obase = Attn + (size_t)b2 * SEQ * EMB + (size_t)h * HD;
#pragma unroll
  for (int rt = 0; rt < 2; ++rt) {
#pragma unroll
    for (int j = 0; j < 8; ++j) {
      float inv = 1.f / l[rt][j];
      size_t R = (size_t)(qr + rt * 16 + j + 8 * hb);
#pragma unroll
      for (int t = 0; t < 4; ++t)
        obase[R * EMB + t * 16 + lo] = (_Float16)(O[rt][t][j] * inv);
    }
  }
}

// ----------------------------------------------- kernel 3: output projection
// Same 4x4 register blocking, f32 output + bias.
__global__ void __launch_bounds__(128)
out_proj_kernel(const _Float16* __restrict__ Ah, const _Float16* __restrict__ Wt,
                const float* __restrict__ bias, float* __restrict__ out) {
  const int NT4 = EMB / 64;                       // 12 column groups
  int wave = blockIdx.x * 4 + (threadIdx.x >> 5); // 64*12 = 768 waves
  int mt4 = wave / NT4, nt4 = wave - mt4 * NT4;
  int lane = threadIdx.x & 31, lo = lane & 15, hb = lane >> 4;
  int kbA = hb * 8, kbB = hb * 16;
  const _Float16* arow = Ah + (size_t)(mt4 * 64 + lo) * EMB;
  const _Float16* bcol = Wt + (size_t)(nt4 * 64 + lo) * EMB;
  v8f zero = {};
  v8f acc[4][4];
#pragma unroll
  for (int mi = 0; mi < 4; ++mi)
#pragma unroll
    for (int ni = 0; ni < 4; ++ni) acc[mi][ni] = zero;

  for (int k0 = 0; k0 < EMB; k0 += 32) {
    v16h a[4], b[4];
#pragma unroll
    for (int mi = 0; mi < 4; ++mi)
      a[mi] = load_a32(arow + (size_t)mi * 16 * EMB + k0, kbA);
#pragma unroll
    for (int ni = 0; ni < 4; ++ni)
      b[ni] = *(const v16h*)(bcol + (size_t)ni * 16 * EMB + k0 + kbB);
#pragma unroll
    for (int mi = 0; mi < 4; ++mi)
#pragma unroll
      for (int ni = 0; ni < 4; ++ni)
        acc[mi][ni] = wmma16(a[mi], b[ni], acc[mi][ni]);
  }

  float bv[4];
#pragma unroll
  for (int ni = 0; ni < 4; ++ni) bv[ni] = bias[nt4 * 64 + ni * 16 + lo];
#pragma unroll
  for (int mi = 0; mi < 4; ++mi)
#pragma unroll
    for (int j = 0; j < 8; ++j) {
      size_t R = (size_t)(mt4 * 64 + mi * 16 + j + 8 * hb);
#pragma unroll
      for (int ni = 0; ni < 4; ++ni)
        out[R * EMB + nt4 * 64 + ni * 16 + lo] = acc[mi][ni][j] + bv[ni];
    }
}

// --------------------------------------------------------------- launcher
extern "C" void kernel_launch(void* const* d_in, const int* in_sizes, int n_in,
                              void* d_out, int out_size, void* d_ws, size_t ws_size,
                              hipStream_t stream) {
  const float* x    = (const float*)d_in[0]; // [2,2048,768]
  const float* Wqkv = (const float*)d_in[1]; // [768,2304]
  const float* bqkv = (const float*)d_in[2]; // [2304]
  const float* Wout = (const float*)d_in[3]; // [768,768]
  const float* bout = (const float*)d_in[4]; // [768]
  float* out = (float*)d_out;                // [2,2048,768] f32

  char* ws = (char*)d_ws;
  size_t off = 0;
  auto alloc = [&](size_t bytes) -> char* {
    char* p = ws + off;
    off += (bytes + 255) & ~(size_t)255;
    return p;
  };
  _Float16* xh   = (_Float16*)alloc((size_t)ROWS * EMB * 2);   // x in f16
  _Float16* wqt  = (_Float16*)alloc((size_t)QKVC * EMB * 2);   // W_qkv^T f16
  _Float16* wot  = (_Float16*)alloc((size_t)EMB * EMB * 2);    // W_out^T f16
  _Float16* Qh   = (_Float16*)alloc((size_t)2 * NH * SEQ * HD * 2);
  _Float16* Kh   = (_Float16*)alloc((size_t)2 * NH * SEQ * HD * 2);
  _Float16* Vth  = (_Float16*)alloc((size_t)2 * NH * HD * SEQ * 2);
  _Float16* attn = (_Float16*)alloc((size_t)ROWS * EMB * 2);
  (void)ws_size; (void)in_sizes; (void)n_in; (void)out_size;

  int nx = ROWS * EMB;
  cvt_f16_kernel<<<(nx + 255) / 256, 256, 0, stream>>>(x, xh, nx);
  cvt_tr_f16_kernel<<<(EMB * QKVC + 255) / 256, 256, 0, stream>>>(Wqkv, wqt, EMB, QKVC);
  cvt_tr_f16_kernel<<<(EMB * EMB + 255) / 256, 256, 0, stream>>>(Wout, wot, EMB, EMB);

  // 64 row groups * 36 col groups = 2304 waves / 4 per block
  qkv_proj_kernel<<<576, 128, 0, stream>>>(xh, wqt, bqkv, Qh, Kh, Vth);
  // 24 (b,h) * 16 q-blocks of 128 rows
  attn_kernel<<<384, 128, 0, stream>>>(Qh, Kh, Vth, attn);
  // 64 row groups * 12 col groups = 768 waves / 4 per block
  out_proj_kernel<<<192, 128, 0, stream>>>(attn, wot, bout, out);
}
